// PredictionModel_21337397527111
// MI455X (gfx1250) — matrix operand order repaired and checked by reading, not compiled, not run on hardware
//
#include <hip/hip_runtime.h>

// Problem constants from the reference
#define BB     16384   // batch
#define DIN    512     // input dim
#define HH     1024    // hidden dim
#define NCODE  8192    // VQ codebook entries

typedef __attribute__((ext_vector_type(16))) __bf16   v16bf;
typedef __attribute__((ext_vector_type(8)))  float    v8f;
typedef __attribute__((ext_vector_type(4)))  unsigned u32x4;
typedef __attribute__((ext_vector_type(4)))  unsigned v4u;
typedef __attribute__((ext_vector_type(8)))  int      v8i;
typedef __attribute__((ext_vector_type(4)))  int      v4i;

union BF16Frag { v16bf v; u32x4 q[2]; };

#if defined(__has_builtin)
#  if __has_builtin(__builtin_amdgcn_tensor_load_to_lds)
#    define USE_TDM 1
#  endif
#endif
#ifndef USE_TDM
#  define USE_TDM 0
#endif

// ---------------------------------------------------------------------------
// f32 -> bf16 elementwise convert
// ---------------------------------------------------------------------------
__global__ void cvt_f32_bf16(const float* __restrict__ src,
                             __bf16* __restrict__ dst, size_t n) {
    size_t i = (size_t)blockIdx.x * blockDim.x + threadIdx.x;
    if (i < n) dst[i] = (__bf16)src[i];
}

// ---------------------------------------------------------------------------
// TDM: DMA one 128(N) x 32(K) bf16 weight tile into LDS, padded so each 64B
// row lands at a 112B stride (28 DWORDs) = conflict-free fragment reads.
// D# layout per CDNA5 ISA ch.8: group0 = {count, lds_addr, global_addr, type=2}
// group1 = {data_size=4B, pad_enable, pad_interval=16dw, pad_amount=12dw,
//           tensor dims/strides, tile dims}. Groups 2/3 (and the extra
//           high-dim/gather words) are zero for a 2D tile.
// This toolchain exposes the 6-arg builtin:
//   (uint32x4, int32x8, int32x4, int32x4, int32x8, i32 cpol)
// ---------------------------------------------------------------------------
#define LDSTRIDE 56   // bf16 elements; 112B row stride in LDS

#if USE_TDM
__device__ __forceinline__ void tdm_load_wtile(const __bf16* W, int K, int N,
                                               int n0, int kk, unsigned ldsAddr) {
    unsigned long long ga =
        (unsigned long long)(size_t)(const void*)(W + (size_t)n0 * K + kk);
    unsigned td0 = (unsigned)(K >> 1);            // tensor dim0 in DWORD units
    unsigned td1 = (unsigned)N;                   // tensor dim1 = rows
    unsigned long long s0 = (unsigned long long)(unsigned)(K >> 1); // row stride (dw)

    v4u g0;
    g0[0] = 1u;                                            // count = 1 descriptor
    g0[1] = ldsAddr;                                       // LDS byte address
    g0[2] = (unsigned)(ga & 0xFFFFFFFFull);                // global addr lo
    g0[3] = (unsigned)((ga >> 32) & 0x1FFFFFFull)          // global addr hi (57b)
          | (2u << 30);                                    // type = 2 ("image")

    v8i g1;
    g1[0] = (int)((2u << 16)      // data_size = 4 bytes
                | (1u << 20)      // pad_enable
                | (3u << 22)      // pad_interval: 16 DWORDs (64B row)
                | (11u << 25));   // pad_amount: 12 DWORDs (48B) -> 112B stride
    g1[1] = (int)((td0 & 0xFFFFu) << 16);                  // tensor_dim0[15:0]
    g1[2] = (int)((td0 >> 16) | ((td1 & 0xFFFFu) << 16));  // td0 hi | td1 lo
    g1[3] = (int)((td1 >> 16) | (16u << 16));              // td1 hi | tile_dim0=16dw
    g1[4] = (int)128;                                      // tile_dim1 = 128 rows
    g1[5] = (int)(unsigned)(s0 & 0xFFFFFFFFull);           // dim0 stride lo
    g1[6] = (int)(unsigned)((s0 >> 32) & 0xFFFFull);       // dim0 stride hi
    g1[7] = 0;

    v4i gz4 = {0, 0, 0, 0};                                // groups 2/3 unused (2D)
    v8i gz8 = {0, 0, 0, 0, 0, 0, 0, 0};                    // extra desc words unused
    __builtin_amdgcn_tensor_load_to_lds(g0, g1, gz4, gz4, gz8, 0);
}
#endif

// ---------------------------------------------------------------------------
// Fused bf16 WMMA GEMM:  out = act(A[M,K] @ W[N,K]^T + bias)
// mode 0: bias + optional leaky-relu, write f32 and/or bf16
// mode 1: VQ — dist = rowNorm[m] + colNorm[n] - 2*acc ; atomicMin 64-bit key
// Workgroup: 256 threads = 8 waves; tile 256(M) x 128(N); K step 32.
// Each wave owns 32 rows (2 A fragments) so every B fragment feeds 2 WMMAs.
// Weight tile double-buffered in LDS, filled by the Tensor Data Mover.
// ---------------------------------------------------------------------------
__global__ __launch_bounds__(256)
void gemm_bf16_wmma(const __bf16* __restrict__ A,
                    const __bf16* __restrict__ W,
                    const float*  __restrict__ bias,
                    float*        __restrict__ outF,
                    __bf16*       __restrict__ outB,
                    const float*  __restrict__ rowNorm,
                    const float*  __restrict__ colNorm,
                    unsigned long long* __restrict__ keys,
                    int M, int N, int K, int act, int mode)
{
    __shared__ __bf16 lw[2][128 * LDSTRIDE];

    const int tid  = threadIdx.x;
    const int wave = tid >> 5;
    const int lane = tid & 31;
    const int hl   = lane >> 4;   // 16-lane half
    const int l16  = lane & 15;
    const int m0   = blockIdx.y * 256 + wave * 32;   // 32 rows per wave
    const int n0   = blockIdx.x * 128;

    const v8f vzero = {0.f, 0.f, 0.f, 0.f, 0.f, 0.f, 0.f, 0.f};
    v8f acc0[8], acc1[8];
#pragma unroll
    for (int i = 0; i < 8; ++i) { acc0[i] = vzero; acc1[i] = vzero; }

#if !USE_TDM
    const int srow = tid >> 1;           // manual-staging fallback
    const int schk = (tid & 1) << 4;
#endif

    int cur = 0;
#if USE_TDM
    const unsigned ldsA0 = (unsigned)(size_t)(void*)&lw[0][0];
    const unsigned ldsA1 = (unsigned)(size_t)(void*)&lw[1][0];
    if (wave == 0) {
        tdm_load_wtile(W, K, N, n0, 0, ldsA0);
        __builtin_amdgcn_s_wait_tensorcnt(0);
    }
    __syncthreads();
#endif

    for (int kk = 0; kk < K; kk += 32) {
#if USE_TDM
        if (wave == 0 && kk + 32 < K)      // overlap next tile DMA with compute
            tdm_load_wtile(W, K, N, n0, kk + 32, cur ? ldsA0 : ldsA1);
#else
        __syncthreads();
        {
            const __bf16* wsrc = W + (size_t)(n0 + srow) * K + kk + schk;
            __bf16* wdst = &lw[0][srow * LDSTRIDE + schk];
            *reinterpret_cast<u32x4*>(wdst)     = *reinterpret_cast<const u32x4*>(wsrc);
            *reinterpret_cast<u32x4*>(wdst + 8) = *reinterpret_cast<const u32x4*>(wsrc + 8);
        }
        __syncthreads();
#endif
        const __bf16* lwc = &lw[cur][0];

        // A fragments (16x32 bf16 ISA layout): lanes 0-15 K 0-7/16-23,
        // lanes 16-31 K 8-15/24-31; two M fragments per wave.
        BF16Frag af0, af1;
        const __bf16* asrc0 = A + (size_t)(m0 + l16) * K + kk + hl * 8;
        const __bf16* asrc1 = asrc0 + (size_t)16 * K;
        af0.q[0] = *reinterpret_cast<const u32x4*>(asrc0);
        af0.q[1] = *reinterpret_cast<const u32x4*>(asrc0 + 16);
        af1.q[0] = *reinterpret_cast<const u32x4*>(asrc1);
        af1.q[1] = *reinterpret_cast<const u32x4*>(asrc1 + 16);
        if (kk + 32 < K) {
            __builtin_prefetch(asrc0 + 32, 0, 1);
            __builtin_prefetch(asrc1 + 32, 0, 1);
        }

#pragma unroll
        for (int nt = 0; nt < 8; ++nt) {
            // B fragment (32x16): lane col = l16; lanes 0-15 K 0-7/16-23,
            // lanes 16-31 K 8-15/24-31 -> from padded LDS row
            BF16Frag bf;
            const __bf16* bsrc = lwc + (nt * 16 + l16) * LDSTRIDE + hl * 8;
            bf.q[0] = *reinterpret_cast<const u32x4*>(bsrc);
            bf.q[1] = *reinterpret_cast<const u32x4*>(bsrc + 16);
            acc0[nt] = __builtin_amdgcn_wmma_f32_16x16x32_bf16(
                false, af0.v, false, bf.v, (short)0, acc0[nt], false, false);
            acc1[nt] = __builtin_amdgcn_wmma_f32_16x16x32_bf16(
                false, af1.v, false, bf.v, (short)0, acc1[nt], false, false);
        }

#if USE_TDM
        if (wave == 0) __builtin_amdgcn_s_wait_tensorcnt(0);
        __syncthreads();
        cur ^= 1;
#endif
    }

    if (mode == 0) {
#pragma unroll
        for (int nt = 0; nt < 8; ++nt) {
            const int n  = n0 + nt * 16 + l16;
            const float bv = bias ? bias[n] : 0.f;
#pragma unroll
            for (int mi = 0; mi < 2; ++mi) {
#pragma unroll
                for (int r = 0; r < 8; ++r) {
                    const int m = m0 + mi * 16 + hl * 8 + r;  // C/D layout
                    float v = (mi ? acc1[nt][r] : acc0[nt][r]) + bv;
                    if (act == 1) v = v > 0.f ? v : 0.01f * v;
                    const size_t idx = (size_t)m * N + n;
                    if (outF) outF[idx] = v;
                    if (outB) outB[idx] = (__bf16)v;
                }
            }
        }
    } else {
        // VQ epilogue: per-lane argmin over the wave's 8 column tiles, then
        // one 64-bit atomicMin per (row, lane-column). Key packs an
        // order-preserving f32 mapping with the index (ties -> lowest index).
#pragma unroll
        for (int mi = 0; mi < 2; ++mi) {
#pragma unroll
            for (int r = 0; r < 8; ++r) {
                const int   m  = m0 + mi * 16 + hl * 8 + r;
                const float rn = rowNorm[m];
                unsigned long long best = ~0ULL;
#pragma unroll
                for (int nt = 0; nt < 8; ++nt) {
                    const int   n = n0 + nt * 16 + l16;
                    const float d = rn + colNorm[n]
                                  - 2.f * (mi ? acc1[nt][r] : acc0[nt][r]);
                    unsigned u = __float_as_uint(d);
                    u = (u & 0x80000000u) ? ~u : (u | 0x80000000u);
                    unsigned long long key =
                        ((unsigned long long)u << 32) | (unsigned)n;
                    best = key < best ? key : best;
                }
                atomicMin(&keys[m], best);
            }
        }
    }
}

// ---------------------------------------------------------------------------
// row-wise sum of squares (f32 input), one wave per row
// ---------------------------------------------------------------------------
__global__ void row_sqnorm(const float* __restrict__ x, float* __restrict__ out,
                           int rows, int cols) {
    int row = blockIdx.x * (blockDim.x >> 5) + (threadIdx.x >> 5);
    if (row >= rows) return;
    int lane = threadIdx.x & 31;
    float s = 0.f;
    for (int c = lane; c < cols; c += 32) {
        float v = x[(size_t)row * cols + c];
        s += v * v;
    }
    for (int off = 16; off > 0; off >>= 1) s += __shfl_down(s, off, 32);
    if (lane == 0) out[row] = s;
}

__global__ void init_keys(unsigned long long* __restrict__ keys, int n) {
    int i = blockIdx.x * blockDim.x + threadIdx.x;
    if (i < n) keys[i] = ~0ULL;
}

// gather codebook row per batch element; emit k (int), z_dec f32, z_dec bf16
__global__ void vq_gather(const unsigned long long* __restrict__ keys,
                          const float* __restrict__ emb,
                          int* __restrict__ kout,
                          float* __restrict__ zdecF,
                          __bf16* __restrict__ zdecB, int cols) {
    int row = blockIdx.x;
    int k = (int)(keys[row] & 0xFFFFFFFFULL);
    if (threadIdx.x == 0) kout[row] = k;
    for (int c = threadIdx.x; c < cols; c += blockDim.x) {
        float v = emb[(size_t)k * cols + c];
        zdecF[(size_t)row * cols + c] = v;
        zdecB[(size_t)row * cols + c] = (__bf16)v;
    }
}

// c[row] = [a[row], b[row]]  (bf16)
__global__ void concat2_bf16(const __bf16* __restrict__ a,
                             const __bf16* __restrict__ b,
                             __bf16* __restrict__ c, int cols) {
    int row = blockIdx.x;
    size_t ro = (size_t)row * 2 * cols;
    for (int cc = threadIdx.x; cc < cols; cc += blockDim.x) {
        c[ro + cc]        = a[(size_t)row * cols + cc];
        c[ro + cols + cc] = b[(size_t)row * cols + cc];
    }
}

// out[row] = tanh(h[row,:] . w + b0), one wave per row
__global__ void gemv_tanh(const __bf16* __restrict__ h,
                          const __bf16* __restrict__ w,
                          const float* __restrict__ b,
                          float* __restrict__ out, int rows, int cols) {
    int row = blockIdx.x * (blockDim.x >> 5) + (threadIdx.x >> 5);
    if (row >= rows) return;
    int lane = threadIdx.x & 31;
    float s = 0.f;
    for (int c = lane; c < cols; c += 32)
        s += (float)h[(size_t)row * cols + c] * (float)w[c];
    for (int off = 16; off > 0; off >>= 1) s += __shfl_down(s, off, 32);
    if (lane == 0) out[row] = tanhf(s + b[0]);
}

__global__ void combine_half_diff(const float* __restrict__ a,
                                  const float* __restrict__ b,
                                  float* __restrict__ o, int n) {
    int i = blockIdx.x * blockDim.x + threadIdx.x;
    if (i < n) o[i] = 0.5f * (a[i] - b[i]);
}

// ---------------------------------------------------------------------------
// launch
// ---------------------------------------------------------------------------
extern "C" void kernel_launch(void* const* d_in, const int* in_sizes, int n_in,
                              void* d_out, int out_size, void* d_ws, size_t ws_size,
                              hipStream_t stream) {
    (void)in_sizes; (void)n_in; (void)out_size; (void)ws_size;

    const float* p1   = (const float*)d_in[0];
    const float* p2   = (const float*)d_in[1];
    const float* ew0  = (const float*)d_in[2];  const float* eb0 = (const float*)d_in[3];
    const float* ew1  = (const float*)d_in[4];  const float* eb1 = (const float*)d_in[5];
    const float* ewo  = (const float*)d_in[6];  const float* ebo = (const float*)d_in[7];
    const float* pw0  = (const float*)d_in[8];  const float* pb0 = (const float*)d_in[9];
    const float* pw1  = (const float*)d_in[10]; const float* pb1 = (const float*)d_in[11];
    const float* pw2  = (const float*)d_in[12]; const float* pb2 = (const float*)d_in[13];
    const float* pw3  = (const float*)d_in[14]; const float* pb3 = (const float*)d_in[15];
    const float* pwo  = (const float*)d_in[16]; const float* pbo = (const float*)d_in[17];
    const float* emb  = (const float*)d_in[18];

    // output layout: x1, z1_dec, x2, z2_dec, k1, k2, v  (flat, in return order)
    const size_t BH = (size_t)BB * HH;
    float* out = (float*)d_out;
    float* x1o = out;
    float* z1o = out + BH;
    float* x2o = out + 2 * BH;
    float* z2o = out + 3 * BH;
    int*   k1o = (int*)(out + 4 * BH);
    int*   k2o = (int*)(out + 4 * BH + BB);
    float* vo  = out + 4 * BH + 2 * (size_t)BB;

    // workspace carving
    char* ws = (char*)d_ws;
    size_t off = 0;
    auto alloc = [&](size_t bytes) -> char* {
        char* p = ws + off;
        off += (bytes + 255) & ~(size_t)255;
        return p;
    };
    __bf16* ew0b = (__bf16*)alloc((size_t)HH * DIN * 2);
    __bf16* ew1b = (__bf16*)alloc((size_t)HH * HH * 2);
    __bf16* ewob = (__bf16*)alloc((size_t)HH * HH * 2);
    __bf16* pw0b = (__bf16*)alloc((size_t)HH * 2 * HH * 2);
    __bf16* pw1b = (__bf16*)alloc((size_t)HH * HH * 2);
    __bf16* pw2b = (__bf16*)alloc((size_t)HH * HH * 2);
    __bf16* pw3b = (__bf16*)alloc((size_t)HH * HH * 2);
    __bf16* pwob = (__bf16*)alloc((size_t)HH * 2);
    __bf16* embb = (__bf16*)alloc((size_t)NCODE * HH * 2);
    __bf16* p1b  = (__bf16*)alloc((size_t)BB * DIN * 2);
    __bf16* p2b  = (__bf16*)alloc((size_t)BB * DIN * 2);
    __bf16* hA   = (__bf16*)alloc((size_t)BB * HH * 2);
    __bf16* hB   = (__bf16*)alloc((size_t)BB * HH * 2);
    __bf16* x1b  = (__bf16*)alloc((size_t)BB * HH * 2);
    __bf16* x2b  = (__bf16*)alloc((size_t)BB * HH * 2);
    __bf16* z1b  = (__bf16*)alloc((size_t)BB * HH * 2);
    __bf16* z2b  = (__bf16*)alloc((size_t)BB * HH * 2);
    __bf16* cbuf = (__bf16*)alloc((size_t)BB * 2 * HH * 2);
    float* en  = (float*)alloc((size_t)NCODE * 4);
    float* zn1 = (float*)alloc((size_t)BB * 4);
    float* zn2 = (float*)alloc((size_t)BB * 4);
    unsigned long long* keys1 = (unsigned long long*)alloc((size_t)BB * 8);
    unsigned long long* keys2 = (unsigned long long*)alloc((size_t)BB * 8);
    float* v1 = (float*)alloc((size_t)BB * 4);
    float* v2 = (float*)alloc((size_t)BB * 4);

    auto cvt = [&](const float* s, __bf16* d, size_t n) {
        cvt_f32_bf16<<<(unsigned)((n + 255) / 256), 256, 0, stream>>>(s, d, n);
    };
    auto gemm = [&](const __bf16* A, const __bf16* W, const float* bias,
                    float* oF, __bf16* oB, int M, int N, int K, int act) {
        dim3 g((unsigned)(N / 128), (unsigned)(M / 256));
        gemm_bf16_wmma<<<g, 256, 0, stream>>>(A, W, bias, oF, oB,
                                              nullptr, nullptr, nullptr,
                                              M, N, K, act, 0);
    };

    // 1) convert weights + inputs to bf16
    cvt(ew0, ew0b, (size_t)HH * DIN);
    cvt(ew1, ew1b, (size_t)HH * HH);
    cvt(ewo, ewob, (size_t)HH * HH);
    cvt(pw0, pw0b, (size_t)HH * 2 * HH);
    cvt(pw1, pw1b, (size_t)HH * HH);
    cvt(pw2, pw2b, (size_t)HH * HH);
    cvt(pw3, pw3b, (size_t)HH * HH);
    cvt(pwo, pwob, (size_t)HH);
    cvt(emb, embb, (size_t)NCODE * HH);
    cvt(p1,  p1b,  (size_t)BB * DIN);
    cvt(p2,  p2b,  (size_t)BB * DIN);

    // 2) encoder (shared weights), x written f32 to d_out + bf16 for VQ
    gemm(p1b, ew0b, eb0, nullptr, hA, BB, HH, DIN, 1);
    gemm(hA,  ew1b, eb1, nullptr, hB, BB, HH, HH, 1);
    gemm(hB,  ewob, ebo, x1o,    x1b, BB, HH, HH, 0);
    gemm(p2b, ew0b, eb0, nullptr, hA, BB, HH, DIN, 1);
    gemm(hA,  ew1b, eb1, nullptr, hB, BB, HH, HH, 1);
    gemm(hB,  ewob, ebo, x2o,    x2b, BB, HH, HH, 0);

    // 3) norms for the distance expansion
    row_sqnorm<<<NCODE / 8, 256, 0, stream>>>(emb, en, NCODE, HH);
    row_sqnorm<<<BB / 8,    256, 0, stream>>>(x1o, zn1, BB, HH);
    row_sqnorm<<<BB / 8,    256, 0, stream>>>(x2o, zn2, BB, HH);

    // 4) fused VQ argmin (never materializes the B x NCODE distance matrix)
    init_keys<<<(BB + 255) / 256, 256, 0, stream>>>(keys1, BB);
    init_keys<<<(BB + 255) / 256, 256, 0, stream>>>(keys2, BB);
    {
        dim3 g(NCODE / 128, BB / 256);
        gemm_bf16_wmma<<<g, 256, 0, stream>>>(x1b, embb, nullptr, nullptr, nullptr,
                                              zn1, en, keys1, BB, NCODE, HH, 0, 1);
        gemm_bf16_wmma<<<g, 256, 0, stream>>>(x2b, embb, nullptr, nullptr, nullptr,
                                              zn2, en, keys2, BB, NCODE, HH, 0, 1);
    }

    // 5) gather codebook rows (k out, z_dec f32 out, z_dec bf16 for predictor)
    vq_gather<<<BB, 256, 0, stream>>>(keys1, emb, k1o, z1o, z1b, HH);
    vq_gather<<<BB, 256, 0, stream>>>(keys2, emb, k2o, z2o, z2b, HH);

    // 6) predictor, direction 1: concat(z1,z2)
    concat2_bf16<<<BB, 256, 0, stream>>>(z1b, z2b, cbuf, HH);
    gemm(cbuf, pw0b, pb0, nullptr, hA, BB, HH, 2 * HH, 1);
    gemm(hA,   pw1b, pb1, nullptr, hB, BB, HH, HH, 1);
    gemm(hB,   pw2b, pb2, nullptr, hA, BB, HH, HH, 1);
    gemm(hA,   pw3b, pb3, nullptr, hB, BB, HH, HH, 1);
    gemv_tanh<<<BB / 8, 256, 0, stream>>>(hB, pwob, pbo, v1, BB, HH);

    // 7) predictor, direction 2: concat(z2,z1)
    concat2_bf16<<<BB, 256, 0, stream>>>(z2b, z1b, cbuf, HH);
    gemm(cbuf, pw0b, pb0, nullptr, hA, BB, HH, 2 * HH, 1);
    gemm(hA,   pw1b, pb1, nullptr, hB, BB, HH, HH, 1);
    gemm(hB,   pw2b, pb2, nullptr, hA, BB, HH, HH, 1);
    gemm(hA,   pw3b, pb3, nullptr, hB, BB, HH, HH, 1);
    gemv_tanh<<<BB / 8, 256, 0, stream>>>(hB, pwob, pbo, v2, BB, HH);

    // 8) v = (v1 - v2) / 2
    combine_half_diff<<<(BB + 255) / 256, 256, 0, stream>>>(v1, v2, vo, BB);
}
